// BinaryLinear_91113436217710
// MI455X (gfx1250) — compile-verified, matrix operand order checked by reading, not run on hardware
//
#include <hip/hip_runtime.h>

typedef __attribute__((ext_vector_type(16))) __bf16 v16bf;
typedef __attribute__((ext_vector_type(8)))  float  v8f;
typedef __attribute__((ext_vector_type(4)))  float  f32x4;   // native vector: OK for nontemporal builtins

#define M_TILE 128
#define N_TILE 128
#define K_TILE 64
#define LDS_K  80   // 64 + 16 pad (keeps 32B alignment of fragment loads, skews banks)

// fp32 -> bf16 round-to-nearest-even, two values packed into one dword
__device__ __forceinline__ unsigned pack_bf16(float a, float b) {
  unsigned ua = __builtin_bit_cast(unsigned, a);
  unsigned ub = __builtin_bit_cast(unsigned, b);
  unsigned lo = (ua + 0x7FFFu + ((ua >> 16) & 1u)) >> 16;
  unsigned hi = (ub + 0x7FFFu + ((ub >> 16) & 1u)) & 0xFFFF0000u;
  return lo | hi;
}

// sign(w) -> bf16 {+1,-1} (zero maps to +1, matching the reference), packed pair
__device__ __forceinline__ unsigned pack_sign(float a, float b) {
  unsigned ua = __builtin_bit_cast(unsigned, a);
  unsigned ub = __builtin_bit_cast(unsigned, b);
  unsigned lo = 0x3F80u | ((ua >> 16) & 0x8000u);
  unsigned hi = 0x3F800000u | (ub & 0x80000000u);
  return lo | hi;
}

__global__ __launch_bounds__(256, 1) void binlin_wmma(
    const float* __restrict__ X, const float* __restrict__ W,
    const float* __restrict__ S, const float* __restrict__ Bias,
    float* __restrict__ Out)
{
  constexpr int M = 8192, N = 16384, K = 4096;
  __shared__ __bf16 lA[M_TILE * LDS_K];
  __shared__ __bf16 lB[N_TILE * LDS_K];

  const int tid  = threadIdx.x;
  const int lane = tid & 31;
  const int wave = tid >> 5;
  // blockIdx.x walks M (fastest): the 64 M-blocks of one N-panel run adjacently
  // and share the 2MB w panel via L2; w is streamed from HBM exactly once.
  // x (128MB) stays L2-resident because w loads and Out stores are non-temporal.
  const int m_blk = blockIdx.x * M_TILE;
  const int n_blk = blockIdx.y * N_TILE;
  // 8 waves arranged 2 (M) x 4 (N); each wave owns a 64x32 output tile
  const int wm = (wave >> 2) * 64;
  const int wn = (wave & 3) * 32;

  v8f acc[4][2];
#pragma unroll
  for (int i = 0; i < 4; ++i)
#pragma unroll
    for (int j = 0; j < 2; ++j)
      acc[i][j] = (v8f){0.f, 0.f, 0.f, 0.f, 0.f, 0.f, 0.f, 0.f};

  // cooperative tile load mapping: 16 float4 per 64-wide row, 16 rows per pass
  const int c4 = (tid & 15) * 4;   // fp32 column within K-tile
  const int r0 = tid >> 4;         // 0..15
  const float* xrow = X + (size_t)(m_blk + r0) * K + c4;
  const float* wrow = W + (size_t)(n_blk + r0) * K + c4;

  for (int k0 = 0; k0 < K; k0 += K_TILE) {
    // ---- stage A tile (x, fp32 -> bf16); regular-temporal: keep x in L2 ----
#pragma unroll
    for (int p = 0; p < 8; ++p) {
      const int r = r0 + p * 16;
      f32x4 v = *(const f32x4*)(xrow + (size_t)(p * 16) * K + k0);
      unsigned* d = (unsigned*)&lA[r * LDS_K + c4];
      d[0] = pack_bf16(v.x, v.y);
      d[1] = pack_bf16(v.z, v.w);
    }
    // ---- stage B tile (w, fp32 -> +-1 bf16); non-temporal: w is used once ----
#pragma unroll
    for (int p = 0; p < 8; ++p) {
      const int r = r0 + p * 16;
      f32x4 v = __builtin_nontemporal_load(
          (const f32x4*)(wrow + (size_t)(p * 16) * K + k0));
      unsigned* d = (unsigned*)&lB[r * LDS_K + c4];
      d[0] = pack_sign(v.x, v.y);
      d[1] = pack_sign(v.z, v.w);
    }
    // prefetch next x K-tile (x is the reuse-heavy stream)
    if (k0 + K_TILE < K) {
      __builtin_prefetch(xrow + k0 + K_TILE, 0, 3);
    }
    __syncthreads();

    // ---- load ALL fragments for both k=32 steps, then 16 back-to-back WMMAs ----
    {
      const int rsel = lane & 15;
      const int klo  = (lane >> 4) << 4;   // lanes 16-31 take K 16..31 of each step
      v16bf a[2][4], b[2][2];
#pragma unroll
      for (int s2 = 0; s2 < 2; ++s2) {
        const int koff = s2 * 32 + klo;
#pragma unroll
        for (int i = 0; i < 4; ++i)
          a[s2][i] = *(const v16bf*)&lA[(wm + i * 16 + rsel) * LDS_K + koff];
#pragma unroll
        for (int j = 0; j < 2; ++j)
          b[s2][j] = *(const v16bf*)&lB[(wn + j * 16 + rsel) * LDS_K + koff];
      }
#pragma unroll
      for (int s2 = 0; s2 < 2; ++s2)
#pragma unroll
        for (int i = 0; i < 4; ++i)
#pragma unroll
          for (int j = 0; j < 2; ++j)
            acc[i][j] = __builtin_amdgcn_wmma_f32_16x16x32_bf16(
                false, a[s2][i], false, b[s2][j], (short)0, acc[i][j], false, false);
    }
    __syncthreads();
  }

  // ---- epilogue: out = scale*acc + bias; non-temporal stores keep L2 for x ----
  const float s = S[0];
#pragma unroll
  for (int i = 0; i < 4; ++i) {
    const int mrow = m_blk + wm + i * 16 + ((lane >> 4) << 3); // +8 for lanes 16-31
#pragma unroll
    for (int j = 0; j < 2; ++j) {
      const int n = n_blk + wn + j * 16 + (lane & 15);
      const float bv = Bias[n];
#pragma unroll
      for (int v = 0; v < 8; ++v)
        __builtin_nontemporal_store(s * acc[i][j][v] + bv,
                                    Out + (size_t)(mrow + v) * N + n);
    }
  }
}

extern "C" void kernel_launch(void* const* d_in, const int* in_sizes, int n_in,
                              void* d_out, int out_size, void* d_ws, size_t ws_size,
                              hipStream_t stream) {
  const float* x    = (const float*)d_in[0];
  const float* w    = (const float*)d_in[1];
  const float* s    = (const float*)d_in[2];
  const float* bias = (const float*)d_in[3];
  float* out = (float*)d_out;

  // grid.x = M blocks (dispatched fastest -> w-panel reuse), grid.y = N blocks
  dim3 grid(8192 / M_TILE, 16384 / N_TILE);  // (64, 128)
  binlin_wmma<<<grid, dim3(256), 0, stream>>>(x, w, s, bias, out);
}
